// Predict_46420006535525
// MI455X (gfx1250) — compile-verified, matrix operand order checked by reading, not run on hardware
//
#include <hip/hip_runtime.h>

typedef __attribute__((ext_vector_type(16))) _Float16 v16h;
typedef __attribute__((ext_vector_type(8)))  float    v8f;

#define TWO_PI 6.283185307179586f
#define DELTA  (TWO_PI/16.0f)
#define INV_SR (1.0f/(0.5625f+1e-8f))   /* 1/(SIG_RHO^2+EPS), SIG_RHO=0.75 */
#define INV_ST (1.0f/(1.0f+1e-8f))
#define EPSV   1e-8f
#define BN_EPS 1e-5f

// VGPR half-index -> K mapping for 16-bit A/B fragments (ISA 7.12.2, 16x32 tile)
__device__ __forceinline__ int kof_map(int h, int lane) {
  return ((h & 8) ? 16 : 0) + ((lane & 16) ? 8 : 0) + (h & 7);
}

// ---------------- weight f32 -> f16 conversion ----------------
__global__ void k_prep_w16(const float* __restrict__ linW, const float* __restrict__ d1W,
                           const float* __restrict__ d2W, _Float16* __restrict__ out) {
  int i = blockIdx.x * blockDim.x + threadIdx.x;
  if (i < 20480)       out[i] = (_Float16)linW[i];
  else if (i < 24576)  out[i] = (_Float16)d1W[i - 20480];
  else if (i < 25600)  out[i] = (_Float16)d2W[i - 24576];
}

// ---------------- fused inference: separable RBF + WMMA GEMM + rot-max ----------------
// one wave32 block per (group, call, b, p); A-tile = 16 rotations x 64 descriptor cols.
// The 64 centers are a separable 4(rho) x 16(theta) grid:
//   exp(-(dr^2/sr + dt^2/st)) = exp(-dr^2/sr) * exp(-dt^2/st)
// rho-exps (x4, mask/feat folded) are rotation-invariant; per rotation only 128x16
// theta-exps are computed cooperatively -> 4x fewer transcendentals than fused form.
__global__ __launch_bounds__(32) void k_inference(
    const float* __restrict__ rho_wt, const float* __restrict__ theta_wt,
    const float* __restrict__ feat_wt, const float* __restrict__ mask_wt,
    const float* __restrict__ rho_mut, const float* __restrict__ theta_mut,
    const float* __restrict__ feat_mut, const float* __restrict__ mask_mut,
    const _Float16* __restrict__ linW16, const float* __restrict__ linb,
    float* __restrict__ gw, float* __restrict__ gm) {
  int blk = blockIdx.x;
  int group = blk / 10240; int jr = blk % 10240;
  int ci = jr / 2048;      int br = jr % 2048;
  int b = br >> 8;         int p = br & 255;
  const float* rho = group ? rho_mut : rho_wt;
  const float* tht = group ? theta_mut : theta_wt;
  const float* fe  = group ? feat_mut : feat_wt;
  const float* ma  = group ? mask_mut : mask_wt;
  float* gout = (group ? gm : gw) + (((b * 5 + ci) * 256 + p) * 64);

  __shared__ float s_th[128];
  __shared__ float s_rexm[512];     // [v][rho-center]  exp(-dr^2/sr)*mask
  __shared__ float s_rexf[512];     // [v][rho-center]  exp(-dr^2/sr)*mask*feat
  __shared__ float s_tth[2048];     // [v][theta-center] per-rotation theta exps
  __shared__ _Float16 s_A[16 * 64];
  __shared__ float s_cf[16 * 64];

  int lane = threadIdx.x;
  const _Float16* Wp = linW16 + ci * 4096;   // [64 out][64 in] row-major
  __builtin_prefetch(Wp, 0, 1);              // global_prefetch_b8: warm weights during exp phase
  __builtin_prefetch(Wp + 2048, 0, 1);

  int base = (b * 256 + p) * 128;
  for (int v = lane; v < 128; v += 32) s_th[v] = tht[base + v];
  for (int idx = lane; idx < 512; idx += 32) {
    int v = idx >> 2, r = idx & 3;
    float dr = rho[base + v] - 1.5f * (float)(r + 1);
    float e = __expf(-dr * dr * INV_SR) * ma[base + v];
    s_rexm[idx] = e;
    s_rexf[idx] = e * fe[(base + v) * 5 + ci];
  }
  __syncthreads();

  int c0 = lane * 2;                // this lane owns descriptor cols c0, c0+1
  int rg = c0 >> 4, t0 = c0 & 15;   // shared rho group, adjacent theta centers

  for (int k = 0; k < 16; ++k) {
    // offset = (ci*120 + k(k+1)/2) * 2pi/16 ; exact mod-2pi via integer &15
    float offr = (float)((ci * 120 + (k * (k + 1)) / 2) & 15) * DELTA;
    for (int idx = lane; idx < 2048; idx += 32) {
      int v = idx >> 4, t = idx & 15;
      float tv = s_th[v] + offr;                     // theta<2pi, offr<2pi
      tv = (tv >= TWO_PI) ? tv - TWO_PI : tv;        // replaces fmodf
      float dt = tv - (float)t * DELTA;
      s_tth[idx] = __expf(-dt * dt * INV_ST);
    }
    __syncthreads();
    float S0 = 0.f, T0 = 0.f, S1 = 0.f, T1 = 0.f;
    for (int v = 0; v < 128; ++v) {
      float a  = s_rexm[v * 4 + rg];
      float f  = s_rexf[v * 4 + rg];
      float e0 = s_tth[v * 16 + t0];
      float e1 = s_tth[v * 16 + t0 + 1];
      S0 += a * e0; T0 += f * e0;
      S1 += a * e1; T1 += f * e1;
    }
    s_A[k * 64 + c0]     = (_Float16)(T0 / (S0 + EPSV));
    s_A[k * 64 + c0 + 1] = (_Float16)(T1 / (S1 + EPSV));
    __syncthreads();   // protects s_tth for next rotation & s_A before GEMM
  }

  // cf[16,64] = A[16,64] @ linW[ci]^T  via v_wmma_f32_16x16x32_f16 (2 K-steps x 4 N-tiles)
  int mrow = lane & 15;
  v8f acc[4] = {};
  for (int kt = 0; kt < 2; ++kt) {
    v16h af;
#pragma unroll
    for (int h = 0; h < 16; ++h) af[h] = s_A[mrow * 64 + kt * 32 + kof_map(h, lane)];
#pragma unroll
    for (int nt = 0; nt < 4; ++nt) {
      v16h bf;
      int n = nt * 16 + mrow;
#pragma unroll
      for (int h = 0; h < 16; ++h) bf[h] = Wp[n * 64 + kt * 32 + kof_map(h, lane)];
      acc[nt] = __builtin_amdgcn_wmma_f32_16x16x32_f16(false, af, false, bf, (short)0,
                                                       acc[nt], false, false);
    }
  }
  int rbase = (lane & 16) ? 8 : 0;
#pragma unroll
  for (int nt = 0; nt < 4; ++nt)
#pragma unroll
    for (int r = 0; r < 8; ++r)
      s_cf[(rbase + r) * 64 + nt * 16 + mrow] = acc[nt][r];
  __syncthreads();

  // max over 16 rotations, + bias, relu
  for (int c = lane * 2; c < lane * 2 + 2; ++c) {
    float mx = -3.4e38f;
    for (int r = 0; r < 16; ++r) mx = fmaxf(mx, s_cf[r * 64 + c]);
    gout[c] = fmaxf(mx + linb[ci * 64 + c], 0.0f);
  }
}

// ---------------- descln: relu(X@W1^T+b1) -> relu(@W2^T+b2), WMMA, 16 rows/wave ----------------
__global__ __launch_bounds__(32) void k_descln(
    const float* __restrict__ X, const _Float16* __restrict__ W1, const float* __restrict__ b1,
    const _Float16* __restrict__ W2, const float* __restrict__ b2, float* __restrict__ Y) {
  __shared__ _Float16 sA[16 * 64];
  __shared__ _Float16 sH[16 * 64];
  int lane = threadIdx.x;
  long row0 = (long)blockIdx.x * 16;
  for (int i = lane; i < 1024; i += 32) sA[i] = (_Float16)X[row0 * 64 + i];
  __syncthreads();
  int mrow = lane & 15;
  int rbase = (lane & 16) ? 8 : 0;
  {
    v8f acc[4] = {};
    for (int kt = 0; kt < 2; ++kt) {
      v16h af;
#pragma unroll
      for (int h = 0; h < 16; ++h) af[h] = sA[mrow * 64 + kt * 32 + kof_map(h, lane)];
#pragma unroll
      for (int nt = 0; nt < 4; ++nt) {
        v16h bf;
        int n = nt * 16 + mrow;
#pragma unroll
        for (int h = 0; h < 16; ++h) bf[h] = W1[n * 64 + kt * 32 + kof_map(h, lane)];
        acc[nt] = __builtin_amdgcn_wmma_f32_16x16x32_f16(false, af, false, bf, (short)0,
                                                         acc[nt], false, false);
      }
    }
#pragma unroll
    for (int nt = 0; nt < 4; ++nt)
#pragma unroll
      for (int r = 0; r < 8; ++r) {
        int col = nt * 16 + mrow;
        sH[(rbase + r) * 64 + col] = (_Float16)fmaxf(acc[nt][r] + b1[col], 0.0f);
      }
  }
  __syncthreads();
  {
    v8f acc = {};
    for (int kt = 0; kt < 2; ++kt) {
      v16h af, bf;
#pragma unroll
      for (int h = 0; h < 16; ++h) {
        int kk = kt * 32 + kof_map(h, lane);
        af[h] = sH[mrow * 64 + kk];
        bf[h] = W2[mrow * 64 + kk];   // W2[16,64], n = mrow
      }
      acc = __builtin_amdgcn_wmma_f32_16x16x32_f16(false, af, false, bf, (short)0, acc, false, false);
    }
#pragma unroll
    for (int r = 0; r < 8; ++r)
      Y[(row0 + rbase + r) * 16 + mrow] = fmaxf(acc[r] + b2[mrow], 0.0f);
  }
}

// ---------------- generic BatchNorm stats / apply ----------------
// addr(e,c) = (e/inner)*(C*inner) + c*inner + e%inner  (inner=1 -> channel-last; inner=HW -> NCHW)
__global__ void k_bn_stats(const float* __restrict__ x, float* __restrict__ mean,
                           float* __restrict__ var, int inner, int C, int Ntot) {
  int c = blockIdx.x, t = threadIdx.x;
  float s = 0.f, s2 = 0.f;
  for (int e = t; e < Ntot; e += 256) {
    int addr = (e / inner) * (C * inner) + c * inner + (e % inner);
    float v = x[addr]; s += v; s2 += v * v;
  }
  __shared__ float rs[256], rq[256];
  rs[t] = s; rq[t] = s2; __syncthreads();
  for (int o = 128; o > 0; o >>= 1) {
    if (t < o) { rs[t] += rs[t + o]; rq[t] += rq[t + o]; }
    __syncthreads();
  }
  if (t == 0) {
    float m = rs[0] / (float)Ntot;
    mean[c] = m;
    var[c]  = rq[0] / (float)Ntot - m * m;
  }
}

__global__ void k_bn_apply(float* __restrict__ x, const float* __restrict__ mean,
                           const float* __restrict__ var, const float* __restrict__ g,
                           const float* __restrict__ bt, int inner, int C, int total) {
  for (int i = blockIdx.x * blockDim.x + threadIdx.x; i < total; i += gridDim.x * blockDim.x) {
    int c = (i / inner) % C;
    x[i] = (x[i] - mean[c]) * (1.0f / sqrtf(var[c] + BN_EPS)) * g[c] + bt[c];
  }
}

// ---------------- diff: inline bnc + |t1-t2| + 1/(t1*t2) + dot(diff_w) + sigmoid + pdm ----------------
__global__ void k_diff(const float* __restrict__ t1x, const float* __restrict__ t2x,
                       const float* __restrict__ m1, const float* __restrict__ v1,
                       const float* __restrict__ m2, const float* __restrict__ v2,
                       const float* __restrict__ bng, const float* __restrict__ bnb,
                       const float* __restrict__ dw, const float* __restrict__ db,
                       const float* __restrict__ pmp, const float* __restrict__ pmask,
                       float* __restrict__ out) {
  int t = blockIdx.x * blockDim.x + threadIdx.x;
  if (t >= 2048) return;
  int b = t >> 8, p = t & 255;
  const float* x1 = t1x + b * 20480 + p * 80;
  const float* x2 = t2x + b * 20480 + p * 80;
  float acc = db[0];
  for (int c = 0; c < 80; ++c) {
    float a1 = (x1[c] - m1[c]) * (1.0f / sqrtf(v1[c] + BN_EPS)) * bng[c] + bnb[c];
    float a2 = (x2[c] - m2[c]) * (1.0f / sqrtf(v2[c] + BN_EPS)) * bng[c] + bnb[c];
    float zs = fabsf(a1 - a2);
    float e1 = (a1 == 0.0f) ? EPSV : a1;
    float e2 = (a2 == 0.0f) ? EPSV : a2;
    acc += a1 * dw[c] + a2 * dw[80 + c] + zs * dw[160 + c] + (1.0f / (e1 * e2)) * dw[240 + c];
  }
  float sg = 1.0f / (1.0f + expf(-acc));
  float s = 0.f;
  for (int q = 0; q < 16; ++q) s += pmp[t * 16 + q];
  out[t] = sg * (s * (1.0f / 16.0f)) * pmask[t];
}

// ---------------- generic FC (8 rows) with optional BN-over-batch + activation ----------------
// act: 0=none 1=relu 2=sigmoid
__global__ __launch_bounds__(64) void k_fc(const float* __restrict__ X, const float* __restrict__ W,
                      const float* __restrict__ bias, const float* __restrict__ gam,
                      const float* __restrict__ bet, float* __restrict__ Y,
                      int K, int F, int use_bn, int act) {
  int f = blockIdx.x, t = threadIdx.x;
  int bb = t & 7, sl = t >> 3;
  float ps = 0.f;
  for (int k = sl; k < K; k += 8) ps += X[bb * K + k] * W[f * K + k];
  __shared__ float red[64];
  __shared__ float yv[8];
  __shared__ float mv[2];
  red[t] = ps; __syncthreads();
  if (t < 8) {
    float y = bias[f];
    for (int s = 0; s < 8; ++s) y += red[t + 8 * s];
    yv[t] = y;
  }
  __syncthreads();
  if (t == 0) {
    float m = 0.f; for (int j = 0; j < 8; ++j) m += yv[j]; m *= 0.125f;
    float v = 0.f; for (int j = 0; j < 8; ++j) { float d = yv[j] - m; v += d * d; } v *= 0.125f;
    mv[0] = m; mv[1] = 1.0f / sqrtf(v + BN_EPS);
  }
  __syncthreads();
  if (t < 8) {
    float y = yv[t];
    if (use_bn) y = (y - mv[0]) * mv[1] * gam[f] + bet[f];
    if (act == 1) y = fmaxf(y, 0.0f);
    else if (act == 2) y = 1.0f / (1.0f + expf(-y));
    Y[t * F + f] = y;
  }
}

// ---------------- dist head input: [dist*mask ; posE raw-reshuffle broadcast] ----------------
__global__ void k_make_distin(const float* __restrict__ dist, const float* __restrict__ dmask,
                              const float* __restrict__ posE, float* __restrict__ out) {
  int i = blockIdx.x * blockDim.x + threadIdx.x;
  if (i >= 8 * 9 * 16384) return;
  int b = i / (9 * 16384); int r = i % (9 * 16384);
  int c = r / 16384;       int yx = r % 16384;
  float v;
  if (c == 0) v = dist[b * 16384 + yx] * dmask[b * 16384 + yx];
  else {
    int flat = (c - 1) * 16384 + yx;       // torch-style raw reshape [128,128,8] -> [8,128,128]
    int ii = flat >> 10, jj = (flat >> 3) & 127, e = flat & 7;
    int rel = ii - jj; rel = rel < -64 ? -64 : (rel > 64 ? 64 : rel);
    v = posE[(rel + 64) * 8 + e];
  }
  out[i] = v;
}

// ---------------- direct 5x5 s2 p2 conv + relu (used for dist head, Cin=9) ----------------
__global__ void k_conv5s2(const float* __restrict__ in, const float* __restrict__ Wt,
                          const float* __restrict__ bias, float* __restrict__ out, int Cin) {
  int i = blockIdx.x * blockDim.x + threadIdx.x;
  if (i >= 8 * 32 * 4096) return;
  int b = i >> 17, o = (i >> 12) & 31, oy = (i >> 6) & 63, ox = i & 63;
  float acc = bias[o];
  for (int c = 0; c < Cin; ++c) {
    const float* ip = in + (((long)b * Cin + c) << 14);
    const float* wp = Wt + (o * Cin + c) * 25;
    for (int ky = 0; ky < 5; ++ky) {
      int y = 2 * oy - 2 + ky; if ((unsigned)y > 127u) continue;
      for (int kx = 0; kx < 5; ++kx) {
        int x = 2 * ox - 2 + kx; if ((unsigned)x > 127u) continue;
        acc += ip[(y << 7) + x] * wp[ky * 5 + kx];
      }
    }
  }
  out[i] = fmaxf(acc, 0.0f);
}

// ---------------- 3x3 s1 p1 conv (32->1) + sigmoid ----------------
__global__ void k_conv3s1(const float* __restrict__ in, const float* __restrict__ Wt,
                          const float* __restrict__ bias, float* __restrict__ out) {
  int i = blockIdx.x * blockDim.x + threadIdx.x;
  if (i >= 8 * 4096) return;
  int b = i >> 12, oy = (i >> 6) & 63, ox = i & 63;
  float acc = bias[0];
  for (int c = 0; c < 32; ++c) {
    const float* ip = in + (((long)b * 32 + c) << 12);
    for (int ky = 0; ky < 3; ++ky) {
      int y = oy - 1 + ky; if ((unsigned)y > 63u) continue;
      for (int kx = 0; kx < 3; ++kx) {
        int x = ox - 1 + kx; if ((unsigned)x > 63u) continue;
        acc += ip[(y << 6) + x] * Wt[(c * 3 + ky) * 3 + kx];
      }
    }
  }
  out[i] = 1.0f / (1.0f + expf(-acc));
}

// ---------------- atom conv1 factorization: P[o,ky,kx,atype] = sum_c E[a,c]*W[o,c,ky,kx] ----------------
__global__ void k_atom_prep(const float* __restrict__ atomE, const float* __restrict__ c1W,
                            float* __restrict__ P1, float* __restrict__ P2) {
  int i = blockIdx.x * blockDim.x + threadIdx.x;
  if (i >= 60800) return;
  int half = i / 30400, r = i % 30400;
  int o = r / 950, rem = r % 950;
  int tap = rem / 38, a = rem % 38;
  int ky = tap / 5, kx = tap % 5;
  float s = 0.f;
  for (int c = 0; c < 64; ++c)
    s += atomE[a * 64 + c] * c1W[(((o * 128) + c + half * 64) * 5 + ky) * 5 + kx];
  (half ? P2 : P1)[((o * 5 + ky) * 5 + kx) * 38 + a] = s;
}

__global__ void k_atom_conv1(const int* __restrict__ awt, const int* __restrict__ amut,
                             const float* __restrict__ P1, const float* __restrict__ P2,
                             const float* __restrict__ bias, float* __restrict__ out) {
  int i = blockIdx.x * blockDim.x + threadIdx.x;
  if (i >= 8 * 32 * 4096) return;
  int b = i >> 17, o = (i >> 12) & 31, oy = (i >> 6) & 63, ox = i & 63;
  float acc = bias[o];
  for (int ky = 0; ky < 5; ++ky) {
    int y = 2 * oy - 2 + ky; if ((unsigned)y > 127u) continue;
    int aw = awt[b * 128 + y];
    for (int kx = 0; kx < 5; ++kx) {
      int x = 2 * ox - 2 + kx; if ((unsigned)x > 127u) continue;
      int am = amut[b * 128 + x];
      int tap = ((o * 5 + ky) * 5 + kx) * 38;
      acc += P1[tap + aw] + P2[tap + am];
    }
  }
  out[i] = fmaxf(acc, 0.0f);
}

__global__ void k_concat(const float* __restrict__ a, const float* __restrict__ b,
                         const float* __restrict__ c, float* __restrict__ out) {
  int i = blockIdx.x * blockDim.x + threadIdx.x;
  if (i >= 1024) return;
  int bb = i >> 7, f = i & 127;
  float v;
  if (f < 64)      v = a[bb * 64 + f];
  else if (f < 96) v = b[bb * 32 + (f - 64)];
  else             v = c[bb * 32 + (f - 96)];
  out[i] = v;
}

// ======================================================================================
extern "C" void kernel_launch(void* const* d_in, const int* in_sizes, int n_in,
                              void* d_out, int out_size, void* d_ws, size_t ws_size,
                              hipStream_t stream) {
  (void)in_sizes; (void)n_in; (void)out_size; (void)ws_size;
  const int PB = 15;  // params dict leaves flattened in insertion order after 15 tensor inputs
  auto P = [&](int i) { return (const float*)d_in[PB + i]; };
  const float* rho_wt   = (const float*)d_in[0];
  const float* theta_wt = (const float*)d_in[1];
  const float* feat_wt  = (const float*)d_in[2];
  const float* mask_wt  = (const float*)d_in[3];
  const float* rho_mut  = (const float*)d_in[4];
  const float* theta_mut= (const float*)d_in[5];
  const float* feat_mut = (const float*)d_in[6];
  const float* mask_mut = (const float*)d_in[7];
  const float* pmask    = (const float*)d_in[9];
  const float* pmp      = (const float*)d_in[10];
  const float* dist     = (const float*)d_in[11];
  const float* dmask    = (const float*)d_in[12];
  const int*   atom_wt  = (const int*)d_in[13];
  const int*   atom_mut = (const int*)d_in[14];

  float* Bp = (float*)d_ws;
  size_t off = 0;
  auto carve = [&](size_t n) { float* q = Bp + off; off += n; return q; };
  float* gw      = carve(655360);     // [8,5,256,64]
  float* gm      = carve(655360);     // contiguous after gw -> 20480x64 rows
  float* gwln    = carve(163840);     // [8,5,256,16] == [8,256,80]
  float* gmln    = carve(163840);     // contiguous after gwln
  float* m1 = carve(80); float* v1 = carve(80); float* m2 = carve(80); float* v2 = carve(80);
  float* diffv   = carve(2048);
  float* dl1o    = carve(1024);
  float* dl2o    = carve(512);
  float* distin  = carve(1179648);    // [8,9,128,128]
  float* dist_a1 = carve(1048576);    // [8,32,64,64]
  float* dist_a2 = carve(32768);      // [8,1,64,64]
  float* dbs_m = carve(32); float* dbs_v = carve(32);
  float* dbs2_m = carve(4); float* dbs2_v = carve(4);
  float* dist_f1 = carve(4096); float* dist_f2 = carve(2048); float* dist_f3 = carve(256);
  float* P1 = carve(30400); float* P2 = carve(30400);
  float* atom_a1 = carve(1048576);
  float* atom_a2 = carve(32768);
  float* abs_m = carve(32); float* abs_v = carve(32);
  float* abs2_m = carve(4); float* abs2_v = carve(4);
  float* atom_f1 = carve(4096); float* atom_f2 = carve(2048); float* atom_f3 = carve(256);
  float* xcat = carve(1024); float* c1o = carve(1024); float* c2o = carve(1024); float* c3o = carve(512);
  _Float16* w16 = (_Float16*)(Bp + off);  // 25600 halves
  const _Float16* linW16 = w16;
  const _Float16* d1W16  = w16 + 20480;
  const _Float16* d2W16  = w16 + 24576;

  k_prep_w16<<<100, 256, 0, stream>>>(P(0), P(2), P(4), w16);

  // 10 inference calls, fused separable-RBF + WMMA GEMM + rotation-max
  k_inference<<<20480, 32, 0, stream>>>(rho_wt, theta_wt, feat_wt, mask_wt,
                                        rho_mut, theta_mut, feat_mut, mask_mut,
                                        linW16, P(1), gw, gm);
  // descln over gw||gm (20480 rows, 16 rows/wave)
  k_descln<<<1280, 32, 0, stream>>>(gw, d1W16, P(3), d2W16, P(5), gwln);

  // bnc stats (channel-last, C=80, N=2048 each)
  k_bn_stats<<<80, 256, 0, stream>>>(gwln, m1, v1, 1, 80, 2048);
  k_bn_stats<<<80, 256, 0, stream>>>(gmln, m2, v2, 1, 80, 2048);
  k_diff<<<8, 256, 0, stream>>>(gwln, gmln, m1, v1, m2, v2, P(8), P(9), P(6), P(7),
                                pmp, pmask, diffv);

  k_fc<<<128, 64, 0, stream>>>(diffv, P(10), P(11), P(12), P(13), dl1o, 256, 128, 1, 1);
  k_fc<<<64, 64, 0, stream>>>(dl1o, P(14), P(15), P(16), P(17), dl2o, 128, 64, 1, 1);

  // ---------- dist head ----------
  k_make_distin<<<4608, 256, 0, stream>>>(dist, dmask, P(18), distin);
  k_conv5s2<<<4096, 256, 0, stream>>>(distin, P(20), P(21), dist_a1, 9);
  k_bn_stats<<<32, 256, 0, stream>>>(dist_a1, dbs_m, dbs_v, 4096, 32, 32768);
  k_bn_apply<<<1024, 256, 0, stream>>>(dist_a1, dbs_m, dbs_v, P(22), P(23), 4096, 32, 1048576);
  k_conv3s1<<<128, 256, 0, stream>>>(dist_a1, P(24), P(25), dist_a2);
  k_bn_stats<<<1, 256, 0, stream>>>(dist_a2, dbs2_m, dbs2_v, 4096, 1, 32768);
  k_bn_apply<<<128, 256, 0, stream>>>(dist_a2, dbs2_m, dbs2_v, P(26), P(27), 4096, 1, 32768);
  k_fc<<<512, 64, 0, stream>>>(dist_a2, P(28), P(29), P(30), P(31), dist_f1, 4096, 512, 1, 1);
  k_fc<<<256, 64, 0, stream>>>(dist_f1, P(32), P(33), P(34), P(35), dist_f2, 512, 256, 1, 1);
  k_fc<<<32, 64, 0, stream>>>(dist_f2, P(36), P(37), P(37), P(37), dist_f3, 256, 32, 0, 1);

  // ---------- atom head (factored conv1: 128x fewer MACs than direct) ----------
  k_atom_prep<<<238, 256, 0, stream>>>(P(19), P(38), P1, P2);
  k_atom_conv1<<<4096, 256, 0, stream>>>(atom_wt, atom_mut, P1, P2, P(39), atom_a1);
  k_bn_stats<<<32, 256, 0, stream>>>(atom_a1, abs_m, abs_v, 4096, 32, 32768);
  k_bn_apply<<<1024, 256, 0, stream>>>(atom_a1, abs_m, abs_v, P(40), P(41), 4096, 32, 1048576);
  k_conv3s1<<<128, 256, 0, stream>>>(atom_a1, P(42), P(43), atom_a2);
  k_bn_stats<<<1, 256, 0, stream>>>(atom_a2, abs2_m, abs2_v, 4096, 1, 32768);
  k_bn_apply<<<128, 256, 0, stream>>>(atom_a2, abs2_m, abs2_v, P(44), P(45), 4096, 1, 32768);
  k_fc<<<512, 64, 0, stream>>>(atom_a2, P(46), P(47), P(48), P(49), atom_f1, 4096, 512, 1, 1);
  k_fc<<<256, 64, 0, stream>>>(atom_f1, P(50), P(51), P(52), P(53), atom_f2, 512, 256, 1, 1);
  k_fc<<<32, 64, 0, stream>>>(atom_f2, P(54), P(55), P(55), P(55), atom_f3, 256, 32, 0, 1);

  // ---------- final classifier ----------
  k_concat<<<4, 256, 0, stream>>>(dl2o, dist_f3, atom_f3, xcat);
  k_fc<<<128, 64, 0, stream>>>(xcat, P(56), P(57), P(58), P(59), c1o, 128, 128, 1, 1);
  k_fc<<<128, 64, 0, stream>>>(c1o, P(60), P(61), P(62), P(63), c2o, 128, 128, 1, 1);
  k_fc<<<64, 64, 0, stream>>>(c2o, P(64), P(65), P(66), P(67), c3o, 128, 64, 1, 1);
  k_fc<<<1, 64, 0, stream>>>(c3o, P(68), P(69), P(69), P(69), (float*)d_out, 64, 1, 0, 2);
}
